// ModulatedConv2D_89953795047583
// MI455X (gfx1250) — compile-verified, matrix operand order checked by reading, not run on hardware
//
#include <hip/hip_runtime.h>

// ---------------------------------------------------------------------------
// StyleGAN2 modulated conv2d for MI455X (gfx1250), bf16 WMMA path +
// async global->LDS double-buffered staging.
// B=8, IN=OUT=512, K=3, RES=64, WDIM=512.
// ---------------------------------------------------------------------------

typedef __bf16 bf16;
typedef bf16  v8bf  __attribute__((ext_vector_type(8)));
typedef bf16  v16bf __attribute__((ext_vector_type(16)));
typedef float v8f   __attribute__((ext_vector_type(8)));
typedef int   v4i   __attribute__((ext_vector_type(4)));

#define NB     8
#define CIN    512
#define COUT   512
#define RES    64
#define WDIM   512
#define PIX    (RES * RES)          // 4096
#define KP     (9 * CIN)            // 4608  (k' = kk*512 + i)
#define PRES   (RES + 2)            // 66 padded
#define CONV_RC 0.014731391f        // 1/sqrt(9*512)
#define DENSE_RC 0.044194173f       // 1/sqrt(512)

// workspace layout (bytes)
#define S_OFF   0
#define D_OFF   (16 * 1024)
#define WB_OFF  (32 * 1024)
#define WB_BYTES ((size_t)NB * COUT * KP * 2)            // 37,748,736
#define XB_OFF  (WB_OFF + WB_BYTES)
#define XB_ELEMS ((size_t)NB * PRES * PRES * CIN)        // 17,842,176
#define XB_BYTES (XB_ELEMS * 2)

// -------- CDNA5 async global->LDS staging (fallback: sync thru VGPRs) ------
#if defined(__HIP_DEVICE_COMPILE__) && defined(__gfx1250__) &&              \
    __has_builtin(__builtin_amdgcn_global_load_async_to_lds_b128) &&        \
    __has_builtin(__builtin_amdgcn_s_wait_asynccnt)
#define HAVE_ASYNC 1
#else
#define HAVE_ASYNC 0
#endif

#if HAVE_ASYNC
typedef __attribute__((address_space(1))) v4i* gas_ptr;
typedef __attribute__((address_space(3))) v4i* lds_ptr;
#endif

__device__ __forceinline__ void stage16(const bf16* g, bf16* l) {
#if HAVE_ASYNC
  __builtin_amdgcn_global_load_async_to_lds_b128((gas_ptr)(void*)(bf16*)g,
                                                 (lds_ptr)(void*)l, 0, 0);
#else
  *(v8bf*)l = *(const v8bf*)g;
#endif
}

__device__ __forceinline__ void wait_stage() {
#if HAVE_ASYNC
  __builtin_amdgcn_s_wait_asynccnt(0);
#endif
}

// ---------------------------------------------------------------------------
// 1) style: s[b,i] = sum_j y[b,j] * w_mod[j,i] * DENSE_RC + b_mod[i] + 1
// ---------------------------------------------------------------------------
__global__ void style_kernel(const float* __restrict__ y,
                             const float* __restrict__ w_mod,
                             const float* __restrict__ b_mod,
                             float* __restrict__ s) {
  const int b = blockIdx.x;
  const int i = threadIdx.x;            // 512 threads
  __shared__ float ys[WDIM];
  ys[i] = y[b * WDIM + i];
  __syncthreads();
  float acc = 0.f;
#pragma unroll 4
  for (int j = 0; j < WDIM; ++j)
    acc += ys[j] * w_mod[j * CIN + i];
  s[b * CIN + i] = acc * DENSE_RC + b_mod[i] + 1.0f;
}

// ---------------------------------------------------------------------------
// 2) demod: d[b,o] = rsqrt(CONV_RC^2 * sum_{kk,i} (w[kk,i,o]*s[b,i])^2 + 1e-8)
// ---------------------------------------------------------------------------
__global__ void demod_kernel(const float* __restrict__ w,
                             const float* __restrict__ s,
                             float* __restrict__ d) {
  const int b = blockIdx.y;
  const int o = blockIdx.x * 256 + threadIdx.x;
  __shared__ float ss[CIN];
  ss[threadIdx.x] = s[b * CIN + threadIdx.x];
  ss[threadIdx.x + 256] = s[b * CIN + threadIdx.x + 256];
  __syncthreads();
  float acc = 0.f;
  for (int kk = 0; kk < 9; ++kk) {
#pragma unroll 4
    for (int i = 0; i < CIN; ++i) {
      float v = w[(kk * CIN + i) * COUT + o] * ss[i];
      acc += v * v;
    }
  }
  d[b * COUT + o] = rsqrtf(acc * (CONV_RC * CONV_RC) + 1e-8f);
}

// ---------------------------------------------------------------------------
// 3) build per-sample weights, transposed to [b][o][k'] bf16 (k' contiguous).
// ---------------------------------------------------------------------------
__global__ void wbuild_kernel(const float* __restrict__ w,
                              const float* __restrict__ s,
                              const float* __restrict__ d,
                              bf16* __restrict__ wb) {
  const int k0 = blockIdx.x * 64;       // 72 tiles over KP
  const int o0 = blockIdx.y * 64;       // 8 tiles over COUT
  const int b  = blockIdx.z;
  const int t  = threadIdx.x;           // 256
  const int kk = k0 / CIN;              // 64-tile never crosses a 512 boundary
  const int ib = k0 % CIN;
  __shared__ bf16 tile[64 * 65];
#pragma unroll
  for (int it = 0; it < 16; ++it) {
    int ol = t & 63;
    int kl = (t >> 6) + it * 4;
    float v = w[(kk * CIN + ib + kl) * COUT + (o0 + ol)] *
              (CONV_RC * s[b * CIN + ib + kl]) * d[b * COUT + o0 + ol];
    tile[ol * 65 + kl] = (bf16)v;
  }
  __syncthreads();
#pragma unroll
  for (int it = 0; it < 16; ++it) {
    int kl = t & 63;
    int ol = (t >> 6) + it * 4;
    wb[((size_t)b * COUT + o0 + ol) * KP + k0 + kl] = tile[ol * 65 + kl];
  }
}

// ---------------------------------------------------------------------------
// 4) zero padded NHWC bf16 buffer (borders must be 0)
// ---------------------------------------------------------------------------
__global__ void xzero_kernel(uint4* __restrict__ xb16) {
  size_t g = (size_t)blockIdx.x * blockDim.x + threadIdx.x;
  xb16[g] = make_uint4(0u, 0u, 0u, 0u);
}

// ---------------------------------------------------------------------------
// 5) x: NCHW fp32 -> padded NHWC bf16.  xb[b][y+1][x+1][i]
// ---------------------------------------------------------------------------
__global__ void xfill_kernel(const float* __restrict__ x,
                             bf16* __restrict__ xb) {
  const int i0 = blockIdx.x * 32;
  const int yy = blockIdx.y;
  const int b  = blockIdx.z;
  const int t  = threadIdx.x;
  __shared__ bf16 tile[64 * 33];
#pragma unroll
  for (int it = 0; it < 8; ++it) {
    int xc = t & 63;
    int il = (t >> 6) + it * 4;
    tile[xc * 33 + il] =
        (bf16)x[((size_t)b * CIN + i0 + il) * PIX + yy * RES + xc];
  }
  __syncthreads();
#pragma unroll
  for (int it = 0; it < 8; ++it) {
    int il = t & 31;
    int xc = (t >> 5) + it * 8;
    xb[(((size_t)b * PRES + yy + 1) * PRES + xc + 1) * CIN + i0 + il] =
        tile[xc * 33 + il];
  }
}

// ---------------------------------------------------------------------------
// 6) main conv = implicit im2col GEMM, V_WMMA_F32_16X16X32_BF16.
//    Block: 256 thr = 8 waves. Tile: 128 o x 128 px (two image rows).
//    Wave w: o-range [o0+16w,+16), 8 n-tiles -> 8 f32 accumulators.
//    Double-buffered async global->LDS staging, 1 barrier / k-step.
// ---------------------------------------------------------------------------
#define LSTR 40
#define NSTEPS 144                     // 9 taps * 16 i-chunks of 32

__global__ __launch_bounds__(256) void conv_kernel(
    const bf16* __restrict__ wb, const bf16* __restrict__ xb,
    float* __restrict__ out) {
  const int y0   = blockIdx.x * 2;     // 32 row-pairs
  const int o0   = blockIdx.y * 128;   // 4 o-tiles
  const int b    = blockIdx.z;         // 8 batches
  const int tid  = threadIdx.x;
  const int wave = tid >> 5;
  const int lane = tid & 31;

  __shared__ __align__(16) bf16 As[2][128 * LSTR];  // 20 KB
  __shared__ __align__(16) bf16 Bs[2][128 * LSTR];  // 20 KB

  const bf16* wbb = wb + ((size_t)b * COUT + o0) * KP;
  const bf16* xbb = xb + (size_t)b * PRES * PRES * CIN;

  // staging: 4 x 16B chunks per thread per step
  const int st_row = tid >> 2;         // 0..63
  const int st_k8  = (tid & 3) * 8;    // 0,8,16,24
  const size_t aog0 = (size_t)st_row * KP + st_k8;
  const size_t aog1 = (size_t)(st_row + 64) * KP + st_k8;
  // pixel p: row r=p>>6, x=p&63  ->  (r*PRES + x)*CIN
  const size_t bog0 = (size_t)(st_row & 63) * CIN + st_k8;          // p<64,r=0
  const size_t bog1 = (size_t)(PRES + (st_row & 63)) * CIN + st_k8; // r=1
  const int al0 = st_row * LSTR + st_k8;
  const int al1 = (st_row + 64) * LSTR + st_k8;

  auto stage = [&](int n, int buf) {
    const int kk = n >> 4;
    const int i0 = (n & 15) * 32;
    const int ky = kk / 3, kx = kk - ky * 3;
    const bf16* wsrc = wbb + kk * CIN + i0;
    const bf16* xsrc = xbb + (size_t)((y0 + ky) * PRES + kx) * CIN + i0;
    stage16(wsrc + aog0, &As[buf][al0]);
    stage16(wsrc + aog1, &As[buf][al1]);
    stage16(xsrc + bog0, &Bs[buf][al0]);
    stage16(xsrc + bog1, &Bs[buf][al1]);
  };

  // fragment LDS offsets (ISA 16-bit A/B per-lane layout)
  const int frow = lane & 15;
  const int fk   = (lane >> 4) * 8;    // K half base; +16 for second half
  const int aoff = (16 * wave + frow) * LSTR + fk;
  const int boff = frow * LSTR + fk;

  v8f acc[8] = {};

  stage(0, 0);
  wait_stage();
  __syncthreads();

  for (int n = 0; n < NSTEPS; ++n) {
    const int buf = n & 1;
    if (n < NSTEPS - 1) stage(n + 1, buf ^ 1);   // prefetch next step

    v8bf alo = *(const v8bf*)&As[buf][aoff];
    v8bf ahi = *(const v8bf*)&As[buf][aoff + 16];
    v16bf a = __builtin_shufflevector(alo, ahi, 0, 1, 2, 3, 4, 5, 6, 7, 8, 9,
                                      10, 11, 12, 13, 14, 15);
#pragma unroll
    for (int h = 0; h < 2; ++h) {                // halves of 4 to cap VGPRs
      v16bf bm[4];
#pragma unroll
      for (int t = 0; t < 4; ++t) {
        int tt = h * 4 + t;
        v8bf blo = *(const v8bf*)&Bs[buf][boff + tt * 16 * LSTR];
        v8bf bhi = *(const v8bf*)&Bs[buf][boff + tt * 16 * LSTR + 16];
        bm[t] = __builtin_shufflevector(blo, bhi, 0, 1, 2, 3, 4, 5, 6, 7, 8,
                                        9, 10, 11, 12, 13, 14, 15);
      }
#pragma unroll
      for (int t = 0; t < 4; ++t)
        acc[h * 4 + t] = __builtin_amdgcn_wmma_f32_16x16x32_bf16(
            false, a, false, bm[t], (short)0, acc[h * 4 + t], false, false);
    }

    wait_stage();      // next buffer's DMA done (own wave's ASYNCcnt)
    __syncthreads();   // everyone done reading buf + staging buf^1
  }

  // store D: lane(0-15) vgpr r -> (M=r,N=lane); lane(16-31) -> (M=8+r,N=lane-16)
  const int mhi  = (lane >> 4) * 8;
  const int ncol = lane & 15;
#pragma unroll
  for (int t = 0; t < 8; ++t) {
#pragma unroll
    for (int r = 0; r < 8; ++r) {
      int orow = o0 + 16 * wave + r + mhi;
      int py   = y0 + (t >> 2);
      int px   = (t & 3) * 16 + ncol;
      out[((size_t)b * COUT + orow) * PIX + py * RES + px] = acc[t][r];
    }
  }
}

// ---------------------------------------------------------------------------
extern "C" void kernel_launch(void* const* d_in, const int* in_sizes, int n_in,
                              void* d_out, int out_size, void* d_ws,
                              size_t ws_size, hipStream_t stream) {
  const float* x     = (const float*)d_in[0];  // [8,512,64,64]
  const float* y     = (const float*)d_in[1];  // [8,512]
  const float* w     = (const float*)d_in[2];  // [3,3,512,512]
  const float* w_mod = (const float*)d_in[3];  // [512,512]
  const float* b_mod = (const float*)d_in[4];  // [512]
  float* out = (float*)d_out;                  // [8,512,64,64]

  char* base = (char*)d_ws;
  float* s  = (float*)(base + S_OFF);
  float* dm = (float*)(base + D_OFF);
  bf16*  wb = (bf16*)(base + WB_OFF);
  bf16*  xb = (bf16*)(base + XB_OFF);

  style_kernel<<<NB, WDIM, 0, stream>>>(y, w_mod, b_mod, s);
  demod_kernel<<<dim3(COUT / 256, NB), 256, 0, stream>>>(w, s, dm);
  wbuild_kernel<<<dim3(KP / 64, COUT / 64, NB), 256, 0, stream>>>(w, s, dm, wb);
  xzero_kernel<<<(unsigned)(XB_BYTES / 16 / 256), 256, 0, stream>>>((uint4*)xb);
  xfill_kernel<<<dim3(CIN / 32, RES, NB), 256, 0, stream>>>(x, xb);
  conv_kernel<<<dim3(RES / 2, COUT / 128, NB), 256, 0, stream>>>(wb, xb, out);
}
